// FastSelfAttention_20117626814931
// MI455X (gfx1250) — matrix operand order repaired and checked by reading, not compiled, not run on hardware
//
#include <hip/hip_runtime.h>

#define N_HEADS 16
#define HEAD    64
#define HID     1024
#define WIN     256
#define LN_EPS  1e-5f

typedef __attribute__((ext_vector_type(16))) _Float16 v16h;
typedef __attribute__((ext_vector_type(8)))  _Float16 v8h;
typedef __attribute__((ext_vector_type(4)))  _Float16 v4h;
typedef __attribute__((ext_vector_type(8)))  float    v8f;

union V16 { v16h v; v8h h[2]; };

// ---------------------------------------------------------------- CDNA5 async global->LDS
__device__ __forceinline__ void async_b128(unsigned lds_off, const void* gptr) {
  asm volatile("global_load_async_to_lds_b128 %0, %1, off"
               :: "v"(lds_off), "v"((unsigned long long)(uintptr_t)gptr)
               : "memory");
}
__device__ __forceinline__ void wait_async0() {
  asm volatile("s_wait_asynccnt 0" ::: "memory");
}

// ---------------------------------------------------------------- f32 -> f16 (x4)
__global__ void cvt_f32_to_f16(const float* __restrict__ x, _Float16* __restrict__ y, int n4) {
  int i = blockIdx.x * blockDim.x + threadIdx.x;
  if (i < n4) {
    float4 f = ((const float4*)x)[i];
    v4h o = { (_Float16)f.x, (_Float16)f.y, (_Float16)f.z, (_Float16)f.w };
    ((v4h*)y)[i] = o;
  }
}

// ---------------------------------------------------------------- LayerNorm
// 256 threads per row of 1024; writes f32 and (optionally) f16.
__global__ __launch_bounds__(256) void ln_kernel(const float* __restrict__ x,
    const float* __restrict__ g, const float* __restrict__ bb,
    float* __restrict__ y32, _Float16* __restrict__ y16) {
  __shared__ float red[16];
  const size_t row = blockIdx.x;
  const float* xr = x + row * HID;
  const int t = threadIdx.x;
  float v[4];
  float s = 0.f, s2 = 0.f;
#pragma unroll
  for (int i = 0; i < 4; ++i) {
    v[i] = xr[t + i * 256];
    s += v[i]; s2 += v[i] * v[i];
  }
#pragma unroll
  for (int o = 16; o > 0; o >>= 1) { s += __shfl_down(s, o, 32); s2 += __shfl_down(s2, o, 32); }
  const int wid = t >> 5, lane = t & 31;
  if (lane == 0) { red[wid] = s; red[8 + wid] = s2; }
  __syncthreads();
  if (t == 0) {
    float a = 0.f, b2 = 0.f;
    for (int i = 0; i < 8; ++i) { a += red[i]; b2 += red[8 + i]; }
    red[0] = a; red[8] = b2;
  }
  __syncthreads();
  const float mean = red[0] * (1.f / HID);
  const float var  = red[8] * (1.f / HID) - mean * mean;
  const float inv  = rsqrtf(var + LN_EPS);
#pragma unroll
  for (int i = 0; i < 4; ++i) {
    const int e = t + i * 256;
    const float o = (v[i] - mean) * inv * g[e] + bb[e];
    y32[row * HID + e] = o;
    if (y16) y16[row * HID + e] = (_Float16)o;
  }
}

// ---------------------------------------------------------------- WMMA GEMM
// C[M,N] = A[M,K](f16) * W[N,K](f16)^T, f32 accumulate.
// Block = 256 threads (8 waves), tile 128x128, BK=32.
// Double-buffered LDS, staged with global_load_async_to_lds_b128 (ASYNCcnt).
#define BM 128
#define BN 128
#define BK 32
#define LDT 40   // LDS stride in halves (80B: padded, 16B aligned)

__global__ __launch_bounds__(256) void wmma_gemm(const _Float16* __restrict__ A,
    const _Float16* __restrict__ W, float* __restrict__ C,
    int M, int N, int K) {
  __shared__ _Float16 sA[2][BM * LDT];
  __shared__ _Float16 sB[2][BN * LDT];

  const int t    = threadIdx.x;
  const int lane = t & 31;
  const int wid  = t >> 5;
  const int wy   = wid & 3;          // 4 wave-rows  (32 rows each)
  const int wx   = wid >> 2;         // 2 wave-cols  (64 cols each)
  const int lr   = lane & 15;        // position within 16
  const int lh   = lane >> 4;        // lane-half select
  const int m0   = blockIdx.y * BM;
  const int n0   = blockIdx.x * BN;
  const int lrow = t >> 1;           // 128 rows, 2 threads/row
  const int lseg = (t & 1) * 16;     // halves

  // Per-lane LDS byte addresses for async staging (flat addr low 32 bits == LDS addr).
  const unsigned aoff = (unsigned)(uintptr_t)(&sA[0][lrow * LDT + lseg]);
  const unsigned boff = (unsigned)(uintptr_t)(&sB[0][lrow * LDT + lseg]);
  const unsigned bufStepA = (unsigned)(sizeof(_Float16) * BM * LDT);
  const unsigned bufStepB = (unsigned)(sizeof(_Float16) * BN * LDT);
  const _Float16* gA = A + (size_t)(m0 + lrow) * K + lseg;
  const _Float16* gB = W + (size_t)(n0 + lrow) * K + lseg;

  auto stage = [&](int buf, int k0) {
    const unsigned ao = aoff + (unsigned)buf * bufStepA;
    const unsigned bo = boff + (unsigned)buf * bufStepB;
    async_b128(ao,      gA + k0);
    async_b128(ao + 16, gA + k0 + 8);
    async_b128(bo,      gB + k0);
    async_b128(bo + 16, gB + k0 + 8);
  };

  v8f zero = {};
  v8f acc[2][4];
#pragma unroll
  for (int mt = 0; mt < 2; ++mt)
#pragma unroll
    for (int nt = 0; nt < 4; ++nt) acc[mt][nt] = zero;

  // Prologue: fill buffer 0.
  stage(0, 0);
  wait_async0();
  __syncthreads();

  int cur = 0;
  for (int k0 = 0; k0 < K; k0 += BK) {
    // Prefetch next tile into the other buffer while computing on this one.
    if (k0 + BK < K) stage(cur ^ 1, k0 + BK);

    V16 af[2], bf[4];
#pragma unroll
    for (int mt = 0; mt < 2; ++mt) {
      // A 16x32 f16: lane<16 -> K[0..7],K[16..23]; lane>=16 -> K[8..15],K[24..31]
      const _Float16* p = &sA[cur][(wy * 32 + mt * 16 + lr) * LDT + lh * 8];
      af[mt].h[0] = *(const v8h*)p;
      af[mt].h[1] = *(const v8h*)(p + 16);
    }
#pragma unroll
    for (int nt = 0; nt < 4; ++nt) {
      // B 32x16 f16: column = lane%16; lane<16 -> K[0..7],K[16..23]; lane>=16 -> K[8..15],K[24..31]
      const _Float16* p = &sB[cur][(wx * 64 + nt * 16 + lr) * LDT + lh * 8];
      bf[nt].h[0] = *(const v8h*)p;
      bf[nt].h[1] = *(const v8h*)(p + 16);
    }
#pragma unroll
    for (int mt = 0; mt < 2; ++mt)
#pragma unroll
      for (int nt = 0; nt < 4; ++nt)
        acc[mt][nt] = __builtin_amdgcn_wmma_f32_16x16x32_f16(
            false, af[mt].v, false, bf[nt].v, (short)0, acc[mt][nt], false, false);

    // Own async writes done, then block-wide visibility.
    wait_async0();
    __syncthreads();
    cur ^= 1;
  }

  // C/D layout: VGPR j -> M = j (lanes 0-15) or j+8 (lanes 16-31); N = lane%16
#pragma unroll
  for (int mt = 0; mt < 2; ++mt)
#pragma unroll
    for (int nt = 0; nt < 4; ++nt) {
      const int n = n0 + wx * 64 + nt * 16 + lr;
      const size_t mbase = (size_t)(m0 + wy * 32 + mt * 16 + lh * 8);
#pragma unroll
      for (int j = 0; j < 8; ++j)
        C[(mbase + j) * N + n] = acc[mt][nt][j];
    }
}

// ---------------------------------------------------------------- per-head logits
// attn[row,h] = exp( dot(X[row,h*64:...], vec[h*64:...]) / 8 + mask_add )
__global__ __launch_bounds__(256) void logits_kernel(const float* __restrict__ X,
    const float* __restrict__ vec, const float* __restrict__ mask,
    float* __restrict__ attn) {
  const size_t row = blockIdx.x;
  const int t = threadIdx.x;
  const float4 q = *(const float4*)(X + row * HID + t * 4);
  const float4 w = *(const float4*)(vec + t * 4);
  float p = q.x * w.x + q.y * w.y + q.z * w.z + q.w * w.w;
#pragma unroll
  for (int o = 8; o > 0; o >>= 1) p += __shfl_down(p, o, 16);
  if ((t & 15) == 0) {
    const float madd = (1.f - mask[row]) * -10000.f;
    attn[row * N_HEADS + (t >> 4)] = __expf(p * 0.125f + madd);
  }
}

// ---------------------------------------------------------------- windowed pooled attention
// out[i] = (P_cur_x[i] + T_prev_x - P_prev_x[i]) / (P_cur_a[i] + T_prev_a - P_prev_a[i])
// One block per (batch, head, chunk of WIN rows); 64 threads = head dims.
__global__ __launch_bounds__(64) void pool_kernel(const float* __restrict__ X,
    const float* __restrict__ attn, float* __restrict__ out, int S) {
  __shared__ float sa[2 * WIN];
  const int nchunks = S / WIN;
  const int c = blockIdx.x % nchunks;
  const int b = blockIdx.x / nchunks;
  const int h = blockIdx.y;
  const size_t row0 = (size_t)b * S + (size_t)c * WIN;
  const int d = threadIdx.x;

  for (int i = d; i < WIN; i += 64) {
    sa[WIN + i] = attn[(row0 + i) * N_HEADS + h];
    sa[i] = (c > 0) ? attn[(row0 - WIN + i) * N_HEADS + h] : 0.f;
  }
  __syncthreads();

  const size_t col = (size_t)h * HEAD + d;
  if (c == 0) {
    float cPa = 0.f, cPx = 0.f;
    for (int i = 0; i < WIN; ++i) {
      const float a = sa[WIN + i];
      cPa += a;
      cPx += a * X[(row0 + i) * HID + col];
      out[(row0 + i) * HID + col] = cPx / cPa;
    }
  } else {
    float Ta = 0.f, Tx = 0.f;
    for (int i = 0; i < WIN; ++i) {
      const float a = sa[i];
      Ta += a;
      Tx += a * X[(row0 - WIN + i) * HID + col];
    }
    float cPa = 0.f, cPx = 0.f, pPa = 0.f, pPx = 0.f;
    for (int i = 0; i < WIN; ++i) {
      const float ac = sa[WIN + i];
      const float ap = sa[i];
      cPa += ac; pPa += ap;
      cPx += ac * X[(row0 + i) * HID + col];
      pPx += ap * X[(row0 - WIN + i) * HID + col];
      out[(row0 + i) * HID + col] = (cPx + Tx - pPx) / (cPa + Ta - pPa);
    }
  }
}

// ---------------------------------------------------------------- out = base + p * m  (x4)
__global__ void fma3_kernel(const float* __restrict__ base, const float* __restrict__ p,
                            const float* __restrict__ m, float* __restrict__ out, int n4) {
  int i = blockIdx.x * blockDim.x + threadIdx.x;
  if (i < n4) {
    float4 a = ((const float4*)base)[i];
    float4 q = ((const float4*)p)[i];
    float4 c = ((const float4*)m)[i];
    float4 o;
    o.x = a.x + q.x * c.x;
    o.y = a.y + q.y * c.y;
    o.z = a.z + q.z * c.z;
    o.w = a.w + q.w * c.w;
    ((float4*)out)[i] = o;
  }
}

// ----------------------------------------------------------------------------
extern "C" void kernel_launch(void* const* d_in, const int* in_sizes, int n_in,
                              void* d_out, int out_size, void* d_ws, size_t ws_size,
                              hipStream_t stream) {
  (void)n_in; (void)out_size; (void)ws_size;
  const float* x   = (const float*)d_in[0];
  const float* msk = (const float*)d_in[1];
  const float* Wq  = (const float*)d_in[2];
  const float* Wk  = (const float*)d_in[3];
  const float* Wv  = (const float*)d_in[4];
  const float* Wp  = (const float*)d_in[5];
  const float* qa  = (const float*)d_in[6];
  const float* ka  = (const float*)d_in[7];
  const float* g1  = (const float*)d_in[8];
  const float* b1  = (const float*)d_in[9];
  const float* g2  = (const float*)d_in[10];
  const float* b2  = (const float*)d_in[11];
  float* out = (float*)d_out;

  const int BS = in_sizes[1];        // B*S (mask)
  const int S  = 4096;               // reference sequence length
  const int B  = BS / S;
  const int M  = BS;                 // GEMM rows
  const int WSZ = HID * HID;         // weight elems
  const int ME  = M * HID;           // activation elems

  char* w = (char*)d_ws;
  auto carve = [&](size_t bytes) { char* p = w; w += (bytes + 255) & ~(size_t)255; return p; };
  _Float16* Wq16 = (_Float16*)carve((size_t)WSZ * 2);
  _Float16* Wk16 = (_Float16*)carve((size_t)WSZ * 2);
  _Float16* Wv16 = (_Float16*)carve((size_t)WSZ * 2);
  _Float16* Wp16 = (_Float16*)carve((size_t)WSZ * 2);
  float*    h32  = (float*)   carve((size_t)ME * 4);
  _Float16* h16  = (_Float16*)carve((size_t)ME * 2);   // also reused for pooled f16
  float*    Qb   = (float*)   carve((size_t)ME * 4);   // Q, then P, then mixed
  float*    Kb   = (float*)   carve((size_t)ME * 4);
  float*    Vb   = (float*)   carve((size_t)ME * 4);
  float*    attnb= (float*)   carve((size_t)M * N_HEADS * 4);
  float*    poolb= (float*)   carve((size_t)ME * 4);   // pooled_q, then pooled_k

  const dim3 ggrid(HID / BN, M / BM);          // 8 x 128 WMMA tiles
  const dim3 pgrid(B * (S / WIN), N_HEADS);    // 64 x 16 pooling blocks
  const int wblk4 = (WSZ / 4 + 255) / 256;
  const int eblk4 = (ME / 4 + 255) / 256;

  // weights -> f16
  cvt_f32_to_f16<<<wblk4, 256, 0, stream>>>(Wq, Wq16, WSZ / 4);
  cvt_f32_to_f16<<<wblk4, 256, 0, stream>>>(Wk, Wk16, WSZ / 4);
  cvt_f32_to_f16<<<wblk4, 256, 0, stream>>>(Wv, Wv16, WSZ / 4);
  cvt_f32_to_f16<<<wblk4, 256, 0, stream>>>(Wp, Wp16, WSZ / 4);

  // h = LN1(x)
  ln_kernel<<<M, 256, 0, stream>>>(x, g1, b1, h32, h16);

  // Q/K/V projections (WMMA, async-LDS staged)
  wmma_gemm<<<ggrid, 256, 0, stream>>>(h16, Wq16, Qb, M, HID, HID);
  wmma_gemm<<<ggrid, 256, 0, stream>>>(h16, Wk16, Kb, M, HID, HID);
  wmma_gemm<<<ggrid, 256, 0, stream>>>(h16, Wv16, Vb, M, HID, HID);

  // pooled_q = additive_attn(Q, query_attn)
  logits_kernel<<<M, 256, 0, stream>>>(Qb, qa, msk, attnb);
  pool_kernel<<<pgrid, 64, 0, stream>>>(Qb, attnb, poolb, S);

  // P = pooled_q @ Wp^T (WMMA); write into Qb (Q dead)
  cvt_f32_to_f16<<<eblk4, 256, 0, stream>>>(poolb, h16, ME / 4);
  wmma_gemm<<<ggrid, 256, 0, stream>>>(h16, Wp16, Qb, M, HID, HID);

  // residual = h + P * keys   (lives in d_out)
  fma3_kernel<<<eblk4, 256, 0, stream>>>(h32, Qb, Kb, out, ME / 4);

  // mixed = LN2(residual) -> Qb (P dead)
  ln_kernel<<<M, 256, 0, stream>>>(out, g2, b2, Qb, (_Float16*)nullptr);

  // pooled_k = additive_attn(mixed, key_attn)
  logits_kernel<<<M, 256, 0, stream>>>(Qb, ka, msk, attnb);
  pool_kernel<<<pgrid, 64, 0, stream>>>(Qb, attnb, poolb, S);

  // out = residual + pooled_k * values (in-place on d_out)
  fma3_kernel<<<eblk4, 256, 0, stream>>>(out, poolb, Vb, out, ME / 4);
}
